// Attention_57080115364607
// MI455X (gfx1250) — matrix operand order repaired and checked by reading, not compiled, not run on hardware
//
#include <hip/hip_runtime.h>
#include <hip/hip_bf16.h>
#include <math.h>

// ---------------------------------------------------------------------------
// Attention layer for MI455X (gfx1250), bf16 WMMA pipeline:
//   k0 : x f32 -> bf16
//   k0b: W f32 [K][N] -> bf16 W^T [N][K]   (Wq|Wk|Wv fused, Wo separate)
//   k1 : fused QKV GEMM (WMMA bf16) + RoPE epilogue; K row-major, V transposed
//   k2 : flash attention (WMMA bf16 QK^T and PV, online softmax)
//   k3 : output projection GEMM (WMMA bf16), f32 result
// Data movers: A tiles via GLOBAL_LOAD_ASYNC_TO_LDS_B128 (ASYNCcnt),
//              B tiles via TENSOR_LOAD_TO_LDS (TDM, TENSORcnt, 6-arg builtin).
// ---------------------------------------------------------------------------

typedef __bf16 bf16;
typedef __attribute__((ext_vector_type(4)))  bf16     v4bf;
typedef __attribute__((ext_vector_type(8)))  bf16     v8bf;
typedef __attribute__((ext_vector_type(16))) bf16     v16bf;
typedef __attribute__((ext_vector_type(4)))  float    v4f;
typedef __attribute__((ext_vector_type(8)))  float    v8f;
typedef __attribute__((ext_vector_type(4)))  int      v4i;
typedef __attribute__((ext_vector_type(8)))  int      v8i;
typedef __attribute__((ext_vector_type(4)))  unsigned v4u;

#define AS1 __attribute__((address_space(1)))
#define AS3 __attribute__((address_space(3)))

#define B_    2
#define S_    2048
#define D_    4096
#define H_    32
#define KVH_  8
#define HD_   128
#define NQKV  (H_*HD_ + 2*KVH_*HD_)   // 6144

#define BM  128
#define BN  128
#define BK  32
#define LDT 40   // LDS tile stride in elements (80B rows, keeps 16B alignment)

#if defined(__has_builtin)
#if __has_builtin(__builtin_amdgcn_global_load_async_to_lds_b128) && \
    __has_builtin(__builtin_amdgcn_s_wait_asynccnt)
#define USE_ASYNC_LDS 1
#endif
#if __has_builtin(__builtin_amdgcn_tensor_load_to_lds) && \
    __has_builtin(__builtin_amdgcn_s_wait_tensorcnt)
#define USE_TDM 1
#endif
#endif
#ifndef USE_ASYNC_LDS
#define USE_ASYNC_LDS 0
#endif
#ifndef USE_TDM
#define USE_TDM 0
#endif

// ---------------------------------------------------------------------------
// WMMA helpers (16x16x32 bf16, f32 accumulate)
// ---------------------------------------------------------------------------
__device__ __forceinline__ v8f wmma_bf16(v16bf a, v16bf b, v8f c) {
  // 8 args: (neg_a, A, neg_b, B, c_mod, C, reuse_a, reuse_b)
  return __builtin_amdgcn_wmma_f32_16x16x32_bf16(false, a, false, b,
                                                 (short)0, c, false, false);
}

__device__ __forceinline__ v16bf frag_cat(v8bf lo, v8bf hi) {
  v16bf r;
#pragma unroll
  for (int i = 0; i < 8; ++i) { r[i] = lo[i]; r[8 + i] = hi[i]; }
  return r;
}

// A fragment: 16(M) x 32(K), source row-major with row stride ld (elements).
// Lanes 0-15 hold M=lane, K 0..15; lanes 16-31 hold K 16..31 (matches B).
__device__ __forceinline__ v16bf load_frag_a(const bf16* p, int lane, int ld) {
  int row = lane & 15;
  int kb  = (lane >> 4) << 4;          // 0 or 16
  const bf16* r0 = p + row * ld + kb;
  v8bf lo = *(const v8bf*)(r0);
  v8bf hi = *(const v8bf*)(r0 + 8);
  return frag_cat(lo, hi);
}

// B fragment: 32(K) x 16(N) from a [N][K] ("B-transposed") source with row
// stride ld. Contiguous k order, same as A: lanes 0-15 hold K=0..15, lanes 16-31 hold K=16..31,
// column N = lane & 15 -> contiguous 16 elements per lane.
__device__ __forceinline__ v16bf load_frag_bt(const bf16* p, int lane, int ld) {
  int col = lane & 15;
  int kb  = (lane >> 4) << 4;          // 0 or 16
  const bf16* r0 = p + col * ld + kb;
  v8bf lo = *(const v8bf*)(r0);
  v8bf hi = *(const v8bf*)(r0 + 8);
  return frag_cat(lo, hi);
}

// ---------------------------------------------------------------------------
// A-tile staging: 128x32 bf16 tile -> LDS (stride LDT), async-to-LDS b128.
// ---------------------------------------------------------------------------
__device__ __forceinline__ void async_cp16(const bf16* src, bf16* dst) {
#if USE_ASYNC_LDS
  __builtin_amdgcn_global_load_async_to_lds_b128(
      (AS1 v4i*)src, (AS3 v4i*)dst, 0, 0);
#else
  *(v8bf*)dst = *(const v8bf*)src;
#endif
}

__device__ __forceinline__ void stage_tile_bf16(bf16* sT, const bf16* A,
                                                int lda, int m0, int k0,
                                                int tid) {
  int row = tid >> 1;
  int c0  = (tid & 1) * 16;
  const bf16* src = A + (size_t)(m0 + row) * lda + k0 + c0;
  bf16* dst = sT + row * LDT + c0;
  async_cp16(src,     dst);
  async_cp16(src + 8, dst + 8);
  __builtin_prefetch((const void*)(src + BK), 0, 1);   // global_prefetch_b8
}

// ---------------------------------------------------------------------------
// B-tile staging: one TDM descriptor moves a 128x32 bf16 tile (row stride
// lda elements) into LDS with LDT=40 element rows via pad_interval/pad_amount
// (64B data + 16B pad per row). Issued by wave 0 only; EXEC is ignored by TDM.
// ---------------------------------------------------------------------------
__device__ __forceinline__ void tdm_load_tile(const bf16* gsrc, bf16* ldst,
                                              int lda) {
#if USE_TDM
  unsigned long long ga = (unsigned long long)(size_t)gsrc;
  unsigned lds = (unsigned)(size_t)(AS3 void*)ldst;
  v4u g0;
  g0[0] = 1u;                                       // count=1, user descriptor
  g0[1] = lds;                                      // lds_addr (bytes)
  g0[2] = (unsigned)(ga & 0xffffffffu);             // global_addr[31:0]
  g0[3] = (unsigned)((ga >> 32) & 0x01ffffffu)      // global_addr[56:32]
        | (2u << 30);                               // type=2 ("image")
  v8i g1;
  g1[0] = (1 << 16)        // data_size = 2 bytes
        | (1 << 20)        // pad_enable
        | (3 << 22)        // pad_interval = 16 DWORDs (64B of data)
        | (3 << 25);       // pad_amount   = 4 DWORDs (16B) -> 80B LDS rows
  g1[1] = (int)(32u << 16);    // tensor_dim0 = 32  (bits 63:48)
  g1[2] = (int)(128u << 16);   // tensor_dim1 = 128 (bits 95:80)
  g1[3] = (int)(32u << 16);    // tile_dim0   = 32  (bits 127:112)
  g1[4] = 128;                 // tile_dim1 = 128, tile_dim2 = 0
  g1[5] = lda;                 // tensor_dim0_stride[31:0] (data_size units)
  g1[6] = 0;                   // stride hi / dim1_stride lo
  g1[7] = 0;
  v4i gz4  = {0, 0, 0, 0};
  v8i gz8  = {0, 0, 0, 0, 0, 0, 0, 0};
  __builtin_amdgcn_tensor_load_to_lds(g0, g1, gz4, gz4, gz8, 0);
#else
  (void)gsrc; (void)ldst; (void)lda;
#endif
}

__device__ __forceinline__ void stage_b_tile(bf16* sT, const bf16* Bsrc,
                                             int n0, int k0, int tid, int wid) {
#if USE_TDM
  if (wid == 0)
    tdm_load_tile(Bsrc + (size_t)n0 * D_ + k0, sT, D_);
  (void)tid;
#else
  stage_tile_bf16(sT, Bsrc, D_, n0, k0, tid);
  (void)wid;
#endif
}

__device__ __forceinline__ void stage_fence_then_barrier() {
#if USE_ASYNC_LDS
  __builtin_amdgcn_s_wait_asynccnt(0);
#endif
#if USE_TDM
  __builtin_amdgcn_s_wait_tensorcnt(0);
#endif
  __syncthreads();
}

__device__ __forceinline__ void mma_step(const bf16* sA, const bf16* sB,
                                         int wm, int wn, int lane,
                                         v8f acc[2][4]) {
  v16bf a0 = load_frag_a(sA + (wm * 32)      * LDT, lane, LDT);
  v16bf a1 = load_frag_a(sA + (wm * 32 + 16) * LDT, lane, LDT);
  v16bf b[4];
#pragma unroll
  for (int ct = 0; ct < 4; ++ct)
    b[ct] = load_frag_bt(sB + (wn * 64 + ct * 16) * LDT, lane, LDT);
#pragma unroll
  for (int ct = 0; ct < 4; ++ct) {
    acc[0][ct] = wmma_bf16(a0, b[ct], acc[0][ct]);
    acc[1][ct] = wmma_bf16(a1, b[ct], acc[1][ct]);
  }
}

// ---------------------------------------------------------------------------
// Kernel 0: x f32 -> bf16
// ---------------------------------------------------------------------------
__global__ void f32_to_bf16_kernel(const float* __restrict__ src,
                                   bf16* __restrict__ dst, int n) {
  int i = (blockIdx.x * blockDim.x + threadIdx.x) * 4;
  if (i < n) {
    v4f v = *(const v4f*)(src + i);
    v4bf o;
#pragma unroll
    for (int j = 0; j < 4; ++j) o[j] = (bf16)v[j];
    *(v4bf*)(dst + i) = o;
  }
}

// ---------------------------------------------------------------------------
// Kernel 0b: W f32 [D_][N] -> W^T bf16 [N][D_]  (64x64 LDS tile transpose)
// ---------------------------------------------------------------------------
#define TT 64
__global__ __launch_bounds__(256) void transpose_cvt_kernel(
    const float* __restrict__ W, bf16* __restrict__ WT, int N) {
  __shared__ alignas(16) bf16 t[TT][TT + 8];
  int n0 = blockIdx.x * TT;
  int k0 = blockIdx.y * TT;
  int tid = threadIdx.x;
  int r  = tid >> 2;            // 0..63
  int c4 = (tid & 3) * 16;      // 0,16,32,48
  const float* src = W + (size_t)(k0 + r) * N + n0 + c4;
#pragma unroll
  for (int i = 0; i < 4; ++i) {
    v4f w = *(const v4f*)(src + 4 * i);
#pragma unroll
    for (int j = 0; j < 4; ++j) t[r][c4 + 4 * i + j] = (bf16)w[j];
  }
  __syncthreads();
  bf16* dst = WT + (size_t)(n0 + r) * D_ + k0 + c4;
  v8bf o0, o1;
#pragma unroll
  for (int j = 0; j < 8; ++j) { o0[j] = t[c4 + j][r]; o1[j] = t[c4 + 8 + j][r]; }
  *(v8bf*)(dst)     = o0;
  *(v8bf*)(dst + 8) = o1;
}

// ---------------------------------------------------------------------------
// Kernel 1: fused QKV projection + RoPE epilogue
//   B operand: qkvT bf16 [6144][4096]  (rows 0..4095 Q | 4096..5119 K | rest V)
//   Q -> [B,H,S,HD] bf16, K -> [B,KVH,S,HD] bf16, V -> [B,KVH,HD,S] bf16
// ---------------------------------------------------------------------------
__global__ __launch_bounds__(256) void qkv_rope_gemm(
    const bf16*  __restrict__ xbf, const bf16* __restrict__ qkvT,
    const float* __restrict__ fc, const float* __restrict__ fs,
    bf16* __restrict__ Qb, bf16* __restrict__ Kb, bf16* __restrict__ Vt) {
  __shared__ alignas(16) bf16 sA[2][BM * LDT];
  __shared__ alignas(16) bf16 sB[2][BN * LDT];

  int tid = threadIdx.x, lane = tid & 31, wid = tid >> 5;
  int wm = wid & 3, wn = wid >> 2;
  int n0 = blockIdx.x * BN;      // global column within fused [Q|K|V] space
  int m0 = blockIdx.y * BM;      // token row (b*S + s)

  int region, nb;                // 128-wide blocks never straddle regions
  if (n0 < H_ * HD_)                    { region = 0; nb = n0; }
  else if (n0 < (H_ + KVH_) * HD_)      { region = 1; nb = n0 - H_ * HD_; }
  else                                  { region = 2; nb = n0 - (H_ + KVH_) * HD_; }

  v8f acc[2][4];
#pragma unroll
  for (int i = 0; i < 2; ++i)
#pragma unroll
    for (int j = 0; j < 4; ++j) acc[i][j] = 0.f;

  stage_tile_bf16(sA[0], xbf, D_, m0, 0, tid);
  stage_b_tile(sB[0], qkvT, n0, 0, tid, wid);

  const int KT = D_ / BK;   // 128 k-steps
  for (int kt = 0; kt < KT; ++kt) {
    stage_fence_then_barrier();
    int cur = kt & 1;
    if (kt + 1 < KT) {
      stage_tile_bf16(sA[cur ^ 1], xbf, D_, m0, (kt + 1) * BK, tid);
      stage_b_tile(sB[cur ^ 1], qkvT, n0, (kt + 1) * BK, tid, wid);
    }
    mma_step(sA[cur], sB[cur], wm, wn, lane, acc);
  }

  // Epilogue: RoPE (Q,K regions) + scatter stores.
  int hi = (lane >> 4) << 3;
  int cb = lane & 15;
#pragma unroll
  for (int rt = 0; rt < 2; ++rt)
#pragma unroll
    for (int ct = 0; ct < 4; ++ct) {
      int ncol = nb + wn * 64 + ct * 16 + cb;   // column within region
#pragma unroll
      for (int r = 0; r < 8; ++r) {
        int mg = m0 + wm * 32 + rt * 16 + r + hi;
        int b  = mg >> 11;           // / S_
        int s  = mg & (S_ - 1);
        float val = acc[rt][ct][r];
        if (region < 2) {            // RoPE: pair (2i,2i+1) lives in lanes l,l^1
          int hd = ncol & (HD_ - 1);
          int pi = hd >> 1;
          float c  = fc[s * (HD_ / 2) + pi];
          float sn = fs[s * (HD_ / 2) + pi];
          float pr = __shfl_xor(val, 1);
          val = (hd & 1) ? (pr * sn + val * c) : (val * c - pr * sn);
        }
        int h  = ncol >> 7;
        int hd = ncol & (HD_ - 1);
        if (region == 0)
          Qb[(((size_t)b * H_ + h) * S_ + s) * HD_ + hd] = (bf16)val;
        else if (region == 1)
          Kb[(((size_t)b * KVH_ + h) * S_ + s) * HD_ + hd] = (bf16)val;
        else
          Vt[(((size_t)b * KVH_ + h) * HD_ + hd) * S_ + s] = (bf16)val;
      }
    }
}

// ---------------------------------------------------------------------------
// Kernel 2: causal flash attention, one 16-row query tile per wave.
// 4 waves / block => 64 query rows per block for one (b,h).
// ---------------------------------------------------------------------------
__global__ __launch_bounds__(128) void flash_attn(
    const bf16* __restrict__ Qb, const bf16* __restrict__ Kb,
    const bf16* __restrict__ Vt, bf16* __restrict__ Ob) {
  __shared__ alignas(16) bf16 sP[4][16 * LDT];   // wave-local P staging

  int tid = threadIdx.x, lane = tid & 31, wid = tid >> 5;
  int bh = blockIdx.y;
  int b = bh >> 5, h = bh & 31;
  int kvh = h >> 2;                 // REP = 4
  int q0 = blockIdx.x * 64 + wid * 16;

  const bf16* Qp = Qb + (((size_t)b * H_ + h) * S_ + q0) * HD_;
  const bf16* Kp = Kb + (((size_t)b * KVH_ + kvh) * S_) * HD_;   // [S][HD]
  const bf16* Vp = Vt + (((size_t)b * KVH_ + kvh) * HD_) * S_;   // [HD][S]

  v16bf qf[4];
#pragma unroll
  for (int ks = 0; ks < 4; ++ks) qf[ks] = load_frag_a(Qp + ks * 32, lane, HD_);

  v8f O[8];
  float m_[8], l_[8];
#pragma unroll
  for (int ct = 0; ct < 8; ++ct) O[ct] = 0.f;
#pragma unroll
  for (int r = 0; r < 8; ++r) { m_[r] = -__builtin_inff(); l_[r] = 0.f; }

  int hi = (lane >> 4) << 3;
  int cb = lane & 15;
  const float sc = 0.08838834764831845f;   // 1/sqrt(HD)

  for (int k0 = 0; k0 < q0 + 16; k0 += 32) {
    // ---- scores: S = Q K^T over HD=128 (4 k-slices), two 16-key tiles ----
    v8f s0 = 0.f, s1 = 0.f;
#pragma unroll
    for (int ks = 0; ks < 4; ++ks) {
      v16bf bk0 = load_frag_bt(Kp + (size_t)(k0)      * HD_ + ks * 32, lane, HD_);
      v16bf bk1 = load_frag_bt(Kp + (size_t)(k0 + 16) * HD_ + ks * 32, lane, HD_);
      s0 = wmma_bf16(qf[ks], bk0, s0);
      s1 = wmma_bf16(qf[ks], bk1, s1);
    }
    // ---- online softmax (half-wave butterfly per C-layout row) ----
    float alpha[8];
#pragma unroll
    for (int r = 0; r < 8; ++r) {
      int row = q0 + r + hi;
      float v0 = s0[r] * sc, v1 = s1[r] * sc;
      if (k0 + cb       > row) v0 = -__builtin_inff();
      if (k0 + 16 + cb  > row) v1 = -__builtin_inff();
      float mx = fmaxf(v0, v1);
#pragma unroll
      for (int d = 1; d < 16; d <<= 1) mx = fmaxf(mx, __shfl_xor(mx, d));
      float mn = fmaxf(m_[r], mx);
      alpha[r] = __expf(m_[r] - mn);
      float p0 = __expf(v0 - mn), p1 = __expf(v1 - mn);
      float sum = p0 + p1;
#pragma unroll
      for (int d = 1; d < 16; d <<= 1) sum += __shfl_xor(sum, d);
      l_[r] = l_[r] * alpha[r] + sum;
      m_[r] = mn;
      sP[wid][(r + hi) * LDT + cb]      = (bf16)p0;
      sP[wid][(r + hi) * LDT + 16 + cb] = (bf16)p1;
    }
#pragma unroll
    for (int ct = 0; ct < 8; ++ct)
#pragma unroll
      for (int r = 0; r < 8; ++r) O[ct][r] *= alpha[r];

    // ---- PV: A = P (16x32 via LDS reshape), B = Vt tiles ----
    v16bf pf = load_frag_a(sP[wid], lane, LDT);
#pragma unroll
    for (int ct = 0; ct < 8; ++ct) {
      v16bf bv = load_frag_bt(Vp + (size_t)(ct * 16) * S_ + k0, lane, S_);
      O[ct] = wmma_bf16(pf, bv, O[ct]);
    }
  }

  // ---- finalize + store attn output [B*S][H*HD] bf16 ----
#pragma unroll
  for (int r = 0; r < 8; ++r) l_[r] = 1.0f / l_[r];
#pragma unroll
  for (int ct = 0; ct < 8; ++ct)
#pragma unroll
    for (int r = 0; r < 8; ++r) {
      int mg  = b * S_ + q0 + r + hi;
      int col = h * HD_ + ct * 16 + cb;
      Ob[(size_t)mg * (H_ * HD_) + col] = (bf16)(O[ct][r] * l_[r]);
    }
}

// ---------------------------------------------------------------------------
// Kernel 3: output projection  out = attn(bf16) @ WoT(bf16), f32 result
// ---------------------------------------------------------------------------
__global__ __launch_bounds__(256) void out_gemm(
    const bf16* __restrict__ Ab, const bf16* __restrict__ WoT,
    float* __restrict__ out) {
  __shared__ alignas(16) bf16 sA[2][BM * LDT];
  __shared__ alignas(16) bf16 sB[2][BN * LDT];

  int tid = threadIdx.x, lane = tid & 31, wid = tid >> 5;
  int wm = wid & 3, wn = wid >> 2;
  int n0 = blockIdx.x * BN;
  int m0 = blockIdx.y * BM;

  v8f acc[2][4];
#pragma unroll
  for (int i = 0; i < 2; ++i)
#pragma unroll
    for (int j = 0; j < 4; ++j) acc[i][j] = 0.f;

  stage_tile_bf16(sA[0], Ab, D_, m0, 0, tid);
  stage_b_tile(sB[0], WoT, n0, 0, tid, wid);

  const int KT = D_ / BK;
  for (int kt = 0; kt < KT; ++kt) {
    stage_fence_then_barrier();
    int cur = kt & 1;
    if (kt + 1 < KT) {
      stage_tile_bf16(sA[cur ^ 1], Ab, D_, m0, (kt + 1) * BK, tid);
      stage_b_tile(sB[cur ^ 1], WoT, n0, (kt + 1) * BK, tid, wid);
    }
    mma_step(sA[cur], sB[cur], wm, wn, lane, acc);
  }

  int hi = (lane >> 4) << 3;
  int cb = lane & 15;
#pragma unroll
  for (int rt = 0; rt < 2; ++rt)
#pragma unroll
    for (int ct = 0; ct < 4; ++ct) {
      int ncol = n0 + wn * 64 + ct * 16 + cb;
#pragma unroll
      for (int r = 0; r < 8; ++r) {
        int mg = m0 + wm * 32 + rt * 16 + r + hi;
        out[(size_t)mg * D_ + ncol] = acc[rt][ct][r];
      }
    }
}

// ---------------------------------------------------------------------------
// Launch
// ---------------------------------------------------------------------------
extern "C" void kernel_launch(void* const* d_in, const int* in_sizes, int n_in,
                              void* d_out, int out_size, void* d_ws, size_t ws_size,
                              hipStream_t stream) {
  (void)in_sizes; (void)n_in; (void)out_size; (void)ws_size;
  const float* x  = (const float*)d_in[0];
  const float* fc = (const float*)d_in[1];
  const float* fs = (const float*)d_in[2];
  const float* Wq = (const float*)d_in[3];
  const float* Wk = (const float*)d_in[4];
  const float* Wv = (const float*)d_in[5];
  const float* Wo = (const float*)d_in[6];
  float* out = (float*)d_out;

  char* ws = (char*)d_ws;
  size_t off = 0;
  auto bump = [&](size_t bytes) {
    char* p = ws + off;
    off += (bytes + 255) & ~(size_t)255;
    return p;
  };
  bf16* xbf  = (bf16*)bump((size_t)B_ * S_ * D_ * 2);
  bf16* qkvT = (bf16*)bump((size_t)NQKV * D_ * 2);          // [6144][4096]
  bf16* WoT  = (bf16*)bump((size_t)D_ * D_ * 2);            // [4096][4096]
  bf16* Qb   = (bf16*)bump((size_t)B_ * H_   * S_ * HD_ * 2);
  bf16* Kb   = (bf16*)bump((size_t)B_ * KVH_ * S_ * HD_ * 2);
  bf16* Vt   = (bf16*)bump((size_t)B_ * KVH_ * S_ * HD_ * 2);
  bf16* Ob   = (bf16*)bump((size_t)B_ * S_ * H_ * HD_ * 2);

  int n = B_ * S_ * D_;
  f32_to_bf16_kernel<<<dim3(n / (256 * 4)), dim3(256), 0, stream>>>(x, xbf, n);

  // Weight transposes into bf16 [N][K] layouts.
  transpose_cvt_kernel<<<dim3((H_*HD_) / TT,   D_ / TT), dim3(256), 0, stream>>>(
      Wq, qkvT, H_ * HD_);
  transpose_cvt_kernel<<<dim3((KVH_*HD_) / TT, D_ / TT), dim3(256), 0, stream>>>(
      Wk, qkvT + (size_t)(H_ * HD_) * D_, KVH_ * HD_);
  transpose_cvt_kernel<<<dim3((KVH_*HD_) / TT, D_ / TT), dim3(256), 0, stream>>>(
      Wv, qkvT + (size_t)((H_ + KVH_) * HD_) * D_, KVH_ * HD_);
  transpose_cvt_kernel<<<dim3(D_ / TT,         D_ / TT), dim3(256), 0, stream>>>(
      Wo, WoT, D_);

  qkv_rope_gemm<<<dim3(NQKV / BN, (B_ * S_) / BM), dim3(256), 0, stream>>>(
      xbf, qkvT, fc, fs, Qb, Kb, Vt);
  flash_attn<<<dim3(S_ / 64, B_ * H_), dim3(128), 0, stream>>>(Qb, Kb, Vt, Ob);
  out_gemm<<<dim3(D_ / BN, (B_ * S_) / BM), dim3(256), 0, stream>>>(Ob, WoT, out);
}
